// CRFRNN_81140522156489
// MI455X (gfx1250) — compile-verified, matrix operand order checked by reading, not run on hardware
//
#include <hip/hip_runtime.h>
#include <hip/hip_bf16.h>
#include <math.h>

// Problem constants (from reference)
constexpr int BB = 16;     // batch
constexpr int NN = 2048;   // sequence length
constexpr int LL = 32;     // labels
constexpr int HH = 768;    // hidden
constexpr int DPP = 64;    // projection dim
constexpr int T_ITERS = 5;

typedef __attribute__((ext_vector_type(16))) __bf16 v16bf;
typedef __attribute__((ext_vector_type(16))) short  v16s;
typedef __attribute__((ext_vector_type(8)))  float  v8f;
typedef __attribute__((ext_vector_type(4)))  unsigned int u32x4;
typedef __attribute__((ext_vector_type(8)))  unsigned int u32x8;

__device__ __forceinline__ unsigned short f2bf(float f) {
  unsigned int u = __builtin_bit_cast(unsigned int, f);
  u += 0x7fffu + ((u >> 16) & 1u);               // round-to-nearest-even
  return (unsigned short)(u >> 16);
}
__device__ __forceinline__ float bf2f(unsigned short s) {
  unsigned int u = ((unsigned int)s) << 16;
  return __builtin_bit_cast(float, u);
}
// Low 32 bits of a generic pointer into the LDS aperture == byte offset in LDS
__device__ __forceinline__ unsigned lds_off(const void* p) {
  return (unsigned)(unsigned long long)(uintptr_t)p;
}

__device__ __forceinline__ v8f wmma_bf16(v16bf a, v16bf b, v8f c) {
  // D = A(16x32 bf16) x B(32x16 bf16) + C(16x16 f32)
  return __builtin_amdgcn_wmma_f32_16x16x32_bf16(
      /*neg_a=*/false, a, /*neg_b=*/false, b,
      /*c_mod=*/(short)0, c, /*reuse_a=*/false, /*reuse_b=*/false);
}

// Load a 16x32 bf16 operand tile in the CDNA5 WMMA A layout from a row-major
// bf16 matrix (row stride `ld` elements). Also used for B tiles by passing the
// transposed operand's rows. Lane L: row m = L&15, half h = L>>4.
//   VGPR v in 0..3 : K = h*8 + 2v, +1        (vector elems 2v, 2v+1)
//   VGPR v in 4..7 : K = 16 + h*8 + 2(v-4)   (vector elems 8+2u, 9+2u)
__device__ __forceinline__ v16bf load_tile_bf16(const unsigned short* base,
                                                int ld, int lane) {
  const int m = lane & 15, h = lane >> 4;
  const unsigned short* row = base + (size_t)m * ld;
  v16s r;
#pragma unroll
  for (int v = 0; v < 4; ++v) {
    unsigned int p0 = *(const unsigned int*)(row + h * 8 + 2 * v);        // 4B-aligned
    r[2 * v]     = (short)(p0 & 0xffffu);
    r[2 * v + 1] = (short)(p0 >> 16);
    unsigned int p1 = *(const unsigned int*)(row + 16 + h * 8 + 2 * v);
    r[8 + 2 * v] = (short)(p1 & 0xffffu);
    r[9 + 2 * v] = (short)(p1 >> 16);
  }
  return __builtin_bit_cast(v16bf, r);
}

// Same layout, source f32 (converted to bf16 on the fly).
__device__ __forceinline__ v16bf load_tile_f32(const float* base, int ld, int lane) {
  const int m = lane & 15, h = lane >> 4;
  const float* row = base + (size_t)m * ld;
  v16s r;
#pragma unroll
  for (int v = 0; v < 4; ++v) {
    const float* p0 = row + h * 8 + 2 * v;
    r[2 * v]     = (short)f2bf(p0[0]);
    r[2 * v + 1] = (short)f2bf(p0[1]);
    const float* p1 = row + 16 + h * 8 + 2 * v;
    r[8 + 2 * v] = (short)f2bf(p1[0]);
    r[9 + 2 * v] = (short)f2bf(p1[1]);
  }
  return __builtin_bit_cast(v16bf, r);
}

// ---------------- Kernel 1: emb = token_reps @ W_emb^T  (bf16 WMMA) ---------
__global__ __launch_bounds__(32)
void proj_kernel(const float* __restrict__ tok, const float* __restrict__ Wemb,
                 unsigned short* __restrict__ emb) {
  const int tile = blockIdx.x;
  const int nt   = blockIdx.y;
  const int lane = threadIdx.x;
  const float* abase = tok  + (size_t)tile * 16 * HH;
  const float* bbase = Wemb + (size_t)nt   * 16 * HH;   // rows of W_emb = B^T rows
  v8f acc = {};
  for (int kk = 0; kk < HH; kk += 32) {
    v16bf a = load_tile_f32(abase + kk, HH, lane);
    v16bf b = load_tile_f32(bbase + kk, HH, lane);
    acc = wmma_bf16(a, b, acc);
  }
  const int n = lane & 15, h = lane >> 4;
#pragma unroll
  for (int r = 0; r < 8; ++r) {
    int row = tile * 16 + r + 8 * h;                    // C layout: m = r + 8*half
    emb[(size_t)row * DPP + nt * 16 + n] = f2bf(acc[r]);
  }
}

// ---------------- Kernel 2: e2[n] = ||emb_n||^2 -----------------------------
__global__ void e2_kernel(const unsigned short* __restrict__ emb,
                          float* __restrict__ e2) {
  int row = blockIdx.x * blockDim.x + threadIdx.x;
  if (row >= BB * NN) return;
  float s = 0.f;
#pragma unroll 8
  for (int d = 0; d < DPP; ++d) { float v = bf2f(emb[(size_t)row * DPP + d]); s += v * v; }
  e2[row] = s;
}

// ---------------- Kernel 3: build row-softmaxed affinity K (bf16) -----------
// One workgroup = (batch b, 16-row slab). TDM stages the emb A-slab into LDS;
// 8 waves compute Gram via WMMA; the full 16x2048 f32 score slab lives in LDS
// (128KB -- CDNA5 320KB WGP LDS) for a single-pass row softmax.
__global__ __launch_bounds__(256)
void buildK_kernel(const unsigned short* __restrict__ emb,
                   const float* __restrict__ e2, const int* __restrict__ mask,
                   const float* __restrict__ wpos_p, const float* __restrict__ wemb_p,
                   const float* __restrict__ lsp_p, const float* __restrict__ lse_p,
                   unsigned short* __restrict__ Kout) {
  extern __shared__ float smem[];
  float* scores = smem;                        // 16*NN f32
  float* red    = smem + 16 * NN;              // 256
  float* rowmax = red + 256;                   // 16
  float* rowinv = rowmax + 16;                 // 16
  unsigned short* A_sh = (unsigned short*)(rowinv + 16);  // 16*DPP bf16 = 2KB

  const int b  = blockIdx.y;
  const int i0 = blockIdx.x * 16;
  const int tid = threadIdx.x;
  const int w = tid >> 5, lane = tid & 31;
  const int n = lane & 15, h = lane >> 4;

  const float wpos = wpos_p[0], wemb = wemb_p[0];
  const float sp = __expf(lsp_p[0]); const float inv2sp2 = 1.f / (2.f * sp * sp);
  const float se = __expf(lse_p[0]); const float inv2se2 = 1.f / (2.f * se * se);

  // ---- TDM: DMA the contiguous 16x64 bf16 A-slab (2KB) into LDS -----------
  const unsigned short* abase = emb + ((size_t)b * NN + i0) * DPP;
  if (w == 0) {
    const unsigned long long ga = (unsigned long long)(uintptr_t)abase;
    u32x4 g0; u32x8 g1;
    g0[0] = 1u;                                   // count=1, user descriptor
    g0[1] = lds_off(A_sh);                        // lds_addr (bytes)
    g0[2] = (unsigned)ga;                         // global_addr[31:0]
    g0[3] = (unsigned)(ga >> 32) | (2u << 30);    // global_addr[56:32] | type=2
    const unsigned elems = 16 * DPP;              // 1024 elements, 1-D copy
    g1[0] = 1u << 16;                             // data_size=1 (2 bytes)
    g1[1] = (elems & 0xffffu) << 16;              // tensor_dim0[15:0]   (bits 63:48)
    g1[2] = (elems >> 16) | (1u << 16);           // dim0[31:16] | tensor_dim1=1
    g1[3] = (elems & 0xffffu) << 16;              // dim1 hi | tile_dim0 (bits 127:112)
    g1[4] = 0u;                                   // tile_dim1=0, tile_dim2=0 (unused)
    g1[5] = elems;                                // tensor_dim0_stride[31:0]
    g1[6] = 0u; g1[7] = 0u;
    asm volatile("tensor_load_to_lds %0, %1" :: "s"(g0), "s"(g1) : "memory");
    __builtin_amdgcn_s_wait_tensorcnt(0);
  }
  __syncthreads();

  // A tile from LDS: 16 rows, K-dim = DP = 64 -> two 32-wide ksteps
  v16bf a0 = load_tile_bf16(A_sh,      DPP, lane);
  v16bf a1 = load_tile_bf16(A_sh + 32, DPP, lane);

  float e2i[8]; int mi[8];
#pragma unroll
  for (int r = 0; r < 8; ++r) {
    e2i[r] = e2[(size_t)b * NN + i0 + r + 8 * h];
    mi[r]  = mask[b * NN + i0 + r + 8 * h];
  }

  for (int jt = w; jt < NN / 16; jt += 8) {
    const int j0 = jt * 16;
    const unsigned short* bbase = emb + ((size_t)b * NN + j0) * DPP;
    if (jt + 8 < NN / 16)
      __builtin_prefetch(bbase + 8 * 16 * DPP, 0, 0);   // global_prefetch_b8
    v16bf b0 = load_tile_bf16(bbase,      DPP, lane);
    v16bf b1 = load_tile_bf16(bbase + 32, DPP, lane);
    v8f acc = {};
    acc = wmma_bf16(a0, b0, acc);
    acc = wmma_bf16(a1, b1, acc);

    const int j   = j0 + n;
    const float e2j = e2[(size_t)b * NN + j];
    const int   mj  = mask[b * NN + j];
#pragma unroll
    for (int r = 0; r < 8; ++r) {
      const int m = r + 8 * h;
      const int i = i0 + m;
      float s;
      if (mi[r] && mj) {
        float dp = (float)(i - j);
        float d2 = e2i[r] + e2j - 2.f * acc[r];
        s = wpos * __expf(-dp * dp * inv2sp2) + wemb * __expf(-d2 * inv2se2);
      } else {
        s = -1e30f;                    // masked (== -inf in reference)
      }
      scores[m * NN + j] = s;
    }
  }
  __syncthreads();

  // row max (16 threads per row)
  { const int r = tid >> 4, c0 = tid & 15;
    float mx = -INFINITY;
    for (int j = c0; j < NN; j += 16) mx = fmaxf(mx, scores[r * NN + j]);
    red[tid] = mx; }
  __syncthreads();
  if (tid < 16) { float mx = -INFINITY;
    for (int k = 0; k < 16; ++k) mx = fmaxf(mx, red[tid * 16 + k]);
    rowmax[tid] = mx; }
  __syncthreads();
  // row sum of exp
  { const int r = tid >> 4, c0 = tid & 15;
    const float rm = rowmax[r]; float s = 0.f;
    for (int j = c0; j < NN; j += 16) { float sc = scores[r * NN + j];
      if (sc > -1e29f) s += __expf(sc - rm); }
    red[tid] = s; }
  __syncthreads();
  if (tid < 16) { float s = 0.f;
    for (int k = 0; k < 16; ++k) s += red[tid * 16 + k];
    rowinv[tid] = (s > 0.f) ? 1.f / s : 0.f; }   // nan_to_num for all-masked rows
  __syncthreads();

  for (int idx = tid; idx < 16 * NN; idx += 256) {
    const int r = idx >> 11, j = idx & (NN - 1);
    const float sc = scores[idx];
    const float v = (sc > -1e29f) ? __expf(sc - rowmax[r]) * rowinv[r] : 0.f;
    Kout[(((size_t)b * NN) + i0 + r) * NN + j] = f2bf(v);
  }
}

// ---------------- Kernel 4: Q0 = softmax(unary), stored transposed (L,N) ----
__global__ void qinit_kernel(const float* __restrict__ unary,
                             unsigned short* __restrict__ Qt) {
  const int idx = blockIdx.x * blockDim.x + threadIdx.x;   // b*N + n
  if (idx >= BB * NN) return;
  const int b = idx / NN, nn = idx % NN;
  const float* u = unary + (size_t)idx * LL;
  float mx = -INFINITY;
#pragma unroll
  for (int l = 0; l < LL; ++l) mx = fmaxf(mx, u[l]);
  float e[LL]; float s = 0.f;
#pragma unroll
  for (int l = 0; l < LL; ++l) { e[l] = __expf(u[l] - mx); s += e[l]; }
  const float inv = 1.f / s;
#pragma unroll
  for (int l = 0; l < LL; ++l)
    Qt[((size_t)b * LL + l) * NN + nn] = f2bf(e[l] * inv);
}

// ---------------- Kernel 5: one mean-field iteration ------------------------
// The 64KB K slab (16 contiguous rows) is staged into LDS with async b128
// copies (ASYNCcnt), then 4 waves run WMMA over LDS A-tiles, split along the
// N=2048 reduction; partials are combined deterministically through LDS.
__global__ __launch_bounds__(128)
void mf_iter_kernel(const unsigned short* __restrict__ Kb,
                    const unsigned short* __restrict__ Qin,
                    const float* __restrict__ unary, const int* __restrict__ mask,
                    const float* __restrict__ mu,
                    unsigned short* __restrict__ Qout, float* __restrict__ out,
                    int final_iter) {
  extern __shared__ char smem_raw[];
  unsigned short* K_sh = (unsigned short*)smem_raw;             // 16*NN bf16 = 64KB
  float* Mpart = (float*)(smem_raw + (size_t)16 * NN * 2);      // [4][16][LL] = 8KB
  float* mu_sh = Mpart + 4 * 16 * LL;                           // 4KB

  const int b  = blockIdx.y;
  const int i0 = blockIdx.x * 16;
  const int tid = threadIdx.x, w = tid >> 5, lane = tid & 31;

  // ---- async-copy the contiguous 16-row K slab into LDS (b128 per lane) ----
  const unsigned short* slab = Kb + ((size_t)b * NN + i0) * NN;
  const unsigned ldsbase = lds_off(K_sh);
  constexpr int SLAB_BYTES = 16 * NN * 2;                       // 65536
#pragma unroll 4
  for (int c = 0; c < SLAB_BYTES / 16 / 128; ++c) {             // 32 iters
    const int off = (c * 128 + tid) * 16;
    const unsigned long long ga = (unsigned long long)(uintptr_t)slab + off;
    const unsigned la = ldsbase + off;
    asm volatile("global_load_async_to_lds_b128 %0, %1, off"
                 :: "v"(la), "v"(ga) : "memory");
  }
  for (int idx = tid; idx < LL * LL; idx += 128) mu_sh[idx] = mu[idx];
  asm volatile("s_wait_asynccnt 0x0" ::: "memory");
  __syncthreads();

  // ---- WMMA message passing: M = K @ Q over this wave's 512-wide k-chunk ---
  const int kbase = w * (NN / 4);
  const unsigned short* b0base = Qin + ((size_t)b * LL + 0 ) * NN + kbase;  // labels 0..15
  const unsigned short* b1base = Qin + ((size_t)b * LL + 16) * NN + kbase;  // labels 16..31
  v8f acc0 = {}, acc1 = {};
  for (int kk = 0; kk < NN / 4; kk += 32) {
    v16bf a   = load_tile_bf16(K_sh + kbase + kk, NN, lane);    // DS gathers
    v16bf bb0 = load_tile_bf16(b0base + kk, NN, lane);
    v16bf bb1 = load_tile_bf16(b1base + kk, NN, lane);
    acc0 = wmma_bf16(a, bb0, acc0);
    acc1 = wmma_bf16(a, bb1, acc1);
  }
  const int n = lane & 15, h = lane >> 4;
#pragma unroll
  for (int r = 0; r < 8; ++r) {
    Mpart[(w * 16 + r + 8 * h) * LL + n]      = acc0[r];
    Mpart[(w * 16 + r + 8 * h) * LL + 16 + n] = acc1[r];
  }
  __syncthreads();

  if (tid < 16) {
    const int i = i0 + tid;
    float Mrow[LL];
#pragma unroll
    for (int l = 0; l < LL; ++l)
      Mrow[l] = Mpart[(0 * 16 + tid) * LL + l] + Mpart[(1 * 16 + tid) * LL + l] +
                Mpart[(2 * 16 + tid) * LL + l] + Mpart[(3 * 16 + tid) * LL + l];
    const float* u = unary + ((size_t)b * NN + i) * LL;
    const int mi = mask[b * NN + i];
    float upd[LL];
#pragma unroll
    for (int k = 0; k < LL; ++k) {
      float S = 0.f;
#pragma unroll
      for (int l = 0; l < LL; ++l) S += Mrow[l] * mu_sh[k * LL + l];
      upd[k] = mi ? (u[k] - S) : -1e9f;
    }
    float mx = -INFINITY;
#pragma unroll
    for (int k = 0; k < LL; ++k) mx = fmaxf(mx, upd[k]);
    float e[LL]; float s = 0.f;
#pragma unroll
    for (int k = 0; k < LL; ++k) { e[k] = __expf(upd[k] - mx); s += e[k]; }
    const float inv = 1.f / s;
    if (final_iter) {
      float* o = out + ((size_t)b * NN + i) * LL;
#pragma unroll
      for (int k = 0; k < LL; ++k) o[k] = __logf(fmaxf(e[k] * inv, 1e-8f));
    } else {
#pragma unroll
      for (int k = 0; k < LL; ++k)
        Qout[((size_t)b * LL + k) * NN + i] = f2bf(e[k] * inv);
    }
  }
}

// ---------------------------- host launcher ---------------------------------
extern "C" void kernel_launch(void* const* d_in, const int* in_sizes, int n_in,
                              void* d_out, int out_size, void* d_ws, size_t ws_size,
                              hipStream_t stream) {
  const float* unary = (const float*)d_in[0];   // (B,N,L)
  const float* tok   = (const float*)d_in[1];   // (B,N,H)
  const int*   mask  = (const int*)  d_in[2];   // (B,N)
  const float* mu    = (const float*)d_in[3];   // (L,L)
  const float* wpos  = (const float*)d_in[4];
  const float* wemb  = (const float*)d_in[5];
  const float* lsp   = (const float*)d_in[6];
  const float* lse   = (const float*)d_in[7];
  const float* Wemb  = (const float*)d_in[8];   // (DP,H)
  float* out = (float*)d_out;

  // workspace layout (bytes)
  char* ws = (char*)d_ws;
  unsigned short* emb = (unsigned short*)ws;                         // 4 MB
  size_t off = (size_t)BB * NN * DPP * 2;
  float* e2 = (float*)(ws + off);                                    // 128 KB
  off += (size_t)BB * NN * 4;
  unsigned short* Qa = (unsigned short*)(ws + off);                  // 2 MB
  off += (size_t)BB * LL * NN * 2;
  unsigned short* Qb = (unsigned short*)(ws + off);                  // 2 MB
  off += (size_t)BB * LL * NN * 2;
  unsigned short* Kb = (unsigned short*)(ws + off);                  // 128 MB

  // 1) projection GEMM (WMMA bf16)
  proj_kernel<<<dim3(BB * NN / 16, DPP / 16), 32, 0, stream>>>(tok, Wemb, emb);
  // 2) squared norms
  e2_kernel<<<(BB * NN + 255) / 256, 256, 0, stream>>>(emb, e2);
  // 3) affinity K: TDM-staged A slab, Gram WMMA, 128KB-LDS one-pass softmax
  const size_t smemK = (size_t)(16 * NN + 256 + 32) * sizeof(float) + 16 * DPP * 2;
  buildK_kernel<<<dim3(NN / 16, BB), 256, smemK, stream>>>(
      emb, e2, mask, wpos, wemb, lsp, lse, Kb);
  // 4) Q init (transposed bf16)
  qinit_kernel<<<(BB * NN + 255) / 256, 256, 0, stream>>>(unary, Qa);
  // 5) T mean-field iterations (async-LDS staged K + WMMA message passing)
  const size_t smemI = (size_t)16 * NN * 2 + (size_t)(4 * 16 * LL + LL * LL) * sizeof(float);
  unsigned short* qi = Qa; unsigned short* qo = Qb;
  for (int t = 0; t < T_ITERS; ++t) {
    mf_iter_kernel<<<dim3(NN / 16, BB), 128, smemI, stream>>>(
        Kb, qi, unary, mask, mu, qo, out, t == T_ITERS - 1 ? 1 : 0);
    unsigned short* tmp = qi; qi = qo; qo = tmp;
  }
}
